// DMSkipGramModel_33466385171083
// MI455X (gfx1250) — matrix operand order repaired and checked by reading, not compiled, not run on hardware
//
#include <hip/hip_runtime.h>
#include <math.h>

#define EMB 128
#define NEG 5
#define NUM_DEP 50

typedef __attribute__((ext_vector_type(16))) _Float16 v16h;
typedef __attribute__((ext_vector_type(8)))  float    v8f;

// ---------------- helpers ----------------
__device__ __forceinline__ float logsig(float x) {
  // stable log(sigmoid(x)) = min(x,0) - log1p(exp(-|x|))
  return fminf(x, 0.0f) - log1pf(expf(-fabsf(x)));
}

// ---------------- phase 0: init scratch ----------------
__global__ void init_kernel(int* bucket, int cap, int* counts, int* cursors,
                            int* tileDep, int maxTiles) {
  int i = blockIdx.x * blockDim.x + threadIdx.x;
  if (i < cap) bucket[i] = -1;
  if (i < NUM_DEP) { counts[i] = 0; cursors[i] = 0; }
  if (i < maxTiles) tileDep[i] = -1;
}

// ---------------- phase 1: histogram of dep labels ----------------
__global__ void count_kernel(const int* __restrict__ dep_label, int B, int* counts) {
  int b = blockIdx.x * blockDim.x + threadIdx.x;
  if (b < B) atomicAdd(&counts[dep_label[b]], 1);
}

// ---------------- phase 2: padded exclusive scan (serial over 50) ----------------
__global__ void scan_kernel(const int* __restrict__ counts, int* poff, int* tileDep) {
  if (threadIdx.x == 0 && blockIdx.x == 0) {
    int run = 0;
    for (int d = 0; d < NUM_DEP; ++d) {
      poff[d] = run;
      int t = (counts[d] + 15) >> 4;       // 16-row tiles for this dep
      int ts = run >> 4;
      for (int i = 0; i < t; ++i) tileDep[ts + i] = d;
      run += t << 4;                        // pad each dep to multiple of 16
    }
  }
}

// ---------------- phase 3: scatter batch ids into dep buckets ----------------
__global__ void scatter_kernel(const int* __restrict__ dep_label, int B,
                               const int* __restrict__ poff, int* cursors, int* bucket) {
  int b = blockIdx.x * blockDim.x + threadIdx.x;
  if (b < B) {
    int d = dep_label[b];
    int p = atomicAdd(&cursors[d], 1);
    bucket[poff[d] + p] = b;
  }
}

// ---------------- phase 3.5: pre-pack D into f16 WMMA B-fragment layout ----------
// Fragment f = n0*4 + kt. Per (dep, f): 32 lanes x v16h, stored contiguously so the
// WMMA kernel reads each lane's fragment as one coalesced 32-byte load.
// bfrag[h] = M[kt*32 + (lane>>4)*8 + (h&7) + (h>>3)*16, n0*16 + (lane&15)]  (B is 32x16 per step)
__global__ __launch_bounds__(32) void pack_m_kernel(const float* __restrict__ D,
                                                    _Float16* __restrict__ M16) {
  const int bid  = blockIdx.x;       // dep*32 + f
  const int dep  = bid >> 5;
  const int f    = bid & 31;
  const int n0   = f >> 2;
  const int kt   = f & 3;
  const int lane = threadIdx.x;
  const int khalf = lane >> 4;
  const int col   = n0 * 16 + (lane & 15);
  const float* __restrict__ Mg = D + (size_t)dep * (EMB * EMB);
  v16h val;
#pragma unroll
  for (int h = 0; h < 16; ++h) {
    int k = kt * 32 + khalf * 8 + (h & 7) + (h >> 3) * 16;
    val[h] = (_Float16)Mg[(size_t)k * EMB + col];
  }
  *(v16h*)(M16 + ((size_t)bid * 512 + lane * 16)) = val;
}

// ---------------- phase 4: WMMA tile kernel ----------------
// One wave32 per tile of 16 batch elements sharing one M = D[dep].
// Z(16x128) = U_tile(16x128) x M(128x128) via v_wmma_f32_16x16x32_f16,
// then per-row dot products against v / noise embeddings.
__global__ __launch_bounds__(32) void wmma_tile_kernel(
    const int* __restrict__ input_label,
    const int* __restrict__ out_label,
    const int* __restrict__ noise,
    const float* __restrict__ U,
    const float* __restrict__ V,
    const int* __restrict__ bucket,
    const int* __restrict__ tileDep,
    const _Float16* __restrict__ M16,
    float* __restrict__ perb)
{
  __shared__ int   rows[16];
  __shared__ float zs[16][132];   // Z tile, padded rows (132 = 33 float4 / row)

  const int tile = blockIdx.x;
  const int dep  = tileDep[tile];
  if (dep < 0) return;            // wave-uniform: EXEC stays all-ones for WMMA

  const int lane = threadIdx.x;   // 0..31 (wave32)
  if (lane < 16) rows[lane] = bucket[tile * 16 + lane];
  __syncthreads();

  // ---- A fragments: rows = u vectors, ISA 16-bit A 16x32 layout ----
  // lane group khalf = lane>>4; per kt the lane needs two runs of 8 consecutive
  // floats: [kt*32 + khalf*8, +8) and [kt*32 + 16 + khalf*8, +8)  -> float4 loads
  const int arow  = lane & 15;
  const int khalf = lane >> 4;
  const int ab    = rows[arow];
  v16h afrag[4];
  if (ab >= 0) {
    const float* __restrict__ uptr = U + (size_t)input_label[ab] * EMB;
#pragma unroll
    for (int kt = 0; kt < 4; ++kt) {
      const int base = kt * 32 + khalf * 8;
      float4 a0 = *(const float4*)(uptr + base);
      float4 a1 = *(const float4*)(uptr + base + 4);
      float4 a2 = *(const float4*)(uptr + base + 16);
      float4 a3 = *(const float4*)(uptr + base + 20);
      afrag[kt][0]  = (_Float16)a0.x; afrag[kt][1]  = (_Float16)a0.y;
      afrag[kt][2]  = (_Float16)a0.z; afrag[kt][3]  = (_Float16)a0.w;
      afrag[kt][4]  = (_Float16)a1.x; afrag[kt][5]  = (_Float16)a1.y;
      afrag[kt][6]  = (_Float16)a1.z; afrag[kt][7]  = (_Float16)a1.w;
      afrag[kt][8]  = (_Float16)a2.x; afrag[kt][9]  = (_Float16)a2.y;
      afrag[kt][10] = (_Float16)a2.z; afrag[kt][11] = (_Float16)a2.w;
      afrag[kt][12] = (_Float16)a3.x; afrag[kt][13] = (_Float16)a3.y;
      afrag[kt][14] = (_Float16)a3.z; afrag[kt][15] = (_Float16)a3.w;
    }
  } else {
#pragma unroll
    for (int kt = 0; kt < 4; ++kt)
#pragma unroll
      for (int h = 0; h < 16; ++h) afrag[kt][h] = (_Float16)0.0f;
  }

  // ---- 8 N-tiles x 4 K-steps of WMMA; B fragments pre-packed in M16 ----
  const v16h* __restrict__ bp = (const v16h*)M16 + (size_t)dep * 1024 + lane;
  const int bcol = lane & 15;
#pragma unroll
  for (int n0 = 0; n0 < 8; ++n0) {
    v8f acc = {};
#pragma unroll
    for (int kt = 0; kt < 4; ++kt) {
      v16h bfrag = bp[(n0 * 4 + kt) * 32];   // one coalesced 32B load per lane
      acc = __builtin_amdgcn_wmma_f32_16x16x32_f16(
          /*neg_a=*/false, afrag[kt], /*neg_b=*/false, bfrag,
          /*c_mod=*/(short)0, acc, /*reuse_a=*/false, /*reuse_b=*/false);
    }
    // C/D layout: VGPR r -> row r + 8*(lane>=16), col = lane&15
#pragma unroll
    for (int r = 0; r < 8; ++r)
      zs[r + khalf * 8][n0 * 16 + bcol] = acc[r];
  }
  __syncthreads();

  // ---- per-row dot products: 2 lanes per row, 64 elements each (float4) ----
  const int row  = lane >> 1;
  const int half = lane & 1;
  const int b    = rows[row];

  float vdp = 0.0f;
  float ndp0 = 0.0f, ndp1 = 0.0f, ndp2 = 0.0f, ndp3 = 0.0f, ndp4 = 0.0f;
  if (b >= 0) {
    const float4* __restrict__ zr  = (const float4*)&zs[row][half * 64];
    const float4* __restrict__ vp  = (const float4*)(V + (size_t)out_label[b] * EMB + half * 64);
    const float4* __restrict__ n0p = (const float4*)(V + (size_t)noise[b * NEG + 0] * EMB + half * 64);
    const float4* __restrict__ n1p = (const float4*)(V + (size_t)noise[b * NEG + 1] * EMB + half * 64);
    const float4* __restrict__ n2p = (const float4*)(V + (size_t)noise[b * NEG + 2] * EMB + half * 64);
    const float4* __restrict__ n3p = (const float4*)(V + (size_t)noise[b * NEG + 3] * EMB + half * 64);
    const float4* __restrict__ n4p = (const float4*)(V + (size_t)noise[b * NEG + 4] * EMB + half * 64);
#pragma unroll 4
    for (int j = 0; j < 16; ++j) {
      float4 z = zr[j];
      float4 a;
      a = vp[j];
      vdp  = fmaf(z.x, a.x, fmaf(z.y, a.y, fmaf(z.z, a.z, fmaf(z.w, a.w, vdp))));
      a = n0p[j];
      ndp0 = fmaf(z.x, a.x, fmaf(z.y, a.y, fmaf(z.z, a.z, fmaf(z.w, a.w, ndp0))));
      a = n1p[j];
      ndp1 = fmaf(z.x, a.x, fmaf(z.y, a.y, fmaf(z.z, a.z, fmaf(z.w, a.w, ndp1))));
      a = n2p[j];
      ndp2 = fmaf(z.x, a.x, fmaf(z.y, a.y, fmaf(z.z, a.z, fmaf(z.w, a.w, ndp2))));
      a = n3p[j];
      ndp3 = fmaf(z.x, a.x, fmaf(z.y, a.y, fmaf(z.z, a.z, fmaf(z.w, a.w, ndp3))));
      a = n4p[j];
      ndp4 = fmaf(z.x, a.x, fmaf(z.y, a.y, fmaf(z.z, a.z, fmaf(z.w, a.w, ndp4))));
    }
  }
  // combine the two half-row partials
  vdp  += __shfl_xor(vdp,  1, 32);
  ndp0 += __shfl_xor(ndp0, 1, 32);
  ndp1 += __shfl_xor(ndp1, 1, 32);
  ndp2 += __shfl_xor(ndp2, 1, 32);
  ndp3 += __shfl_xor(ndp3, 1, 32);
  ndp4 += __shfl_xor(ndp4, 1, 32);

  if (b >= 0 && half == 0) {
    // neg embeddings are negated in the reference: neg_dot = -(z . V[noise])
    float loss = logsig(vdp)
               + logsig(-ndp0) + logsig(-ndp1) + logsig(-ndp2)
               + logsig(-ndp3) + logsig(-ndp4);
    perb[b] = loss;
  }
}

// ---------------- phase 5: deterministic fixed-order reduction ----------------
__global__ void reduce_kernel(const float* __restrict__ perb, int B, float* out) {
  __shared__ float s[256];
  int t = threadIdx.x;
  float acc = 0.0f;
  for (int i = t; i < B; i += 256) acc += perb[i];
  s[t] = acc;
  __syncthreads();
  for (int off = 128; off > 0; off >>= 1) {
    if (t < off) s[t] += s[t + off];
    __syncthreads();
  }
  if (t == 0) out[0] = -s[0] / (float)B;
}

// ---------------- launch ----------------
extern "C" void kernel_launch(void* const* d_in, const int* in_sizes, int n_in,
                              void* d_out, int out_size, void* d_ws, size_t ws_size,
                              hipStream_t stream) {
  const int*   input_label = (const int*)d_in[0];
  const int*   out_label   = (const int*)d_in[1];
  const int*   dep_label   = (const int*)d_in[2];
  const int*   noise       = (const int*)d_in[3];
  const float* U           = (const float*)d_in[4];
  const float* V           = (const float*)d_in[5];
  const float* D           = (const float*)d_in[6];

  const int B        = in_sizes[0];
  const int maxTiles = (B + 15) / 16 + NUM_DEP;  // worst-case padded tile count
  const int cap      = maxTiles * 16;

  // workspace layout
  float* perb    = (float*)d_ws;        // [B]
  int*   bucket  = (int*)(perb + B);    // [cap]
  int*   counts  = bucket + cap;        // [NUM_DEP]
  int*   cursors = counts + NUM_DEP;    // [NUM_DEP]
  int*   poff    = cursors + NUM_DEP;   // [NUM_DEP]
  int*   tileDep = poff + NUM_DEP;      // [maxTiles]
  size_t off = (size_t)(B + cap + 3 * NUM_DEP + maxTiles) * 4;
  off = (off + 255) & ~(size_t)255;     // 256B-align the f16 matrix pack
  _Float16* M16 = (_Float16*)((char*)d_ws + off);  // [NUM_DEP * 32 * 512] halves

  init_kernel<<<(cap + 255) / 256, 256, 0, stream>>>(bucket, cap, counts, cursors,
                                                     tileDep, maxTiles);
  count_kernel<<<(B + 255) / 256, 256, 0, stream>>>(dep_label, B, counts);
  scan_kernel<<<1, 64, 0, stream>>>(counts, poff, tileDep);
  scatter_kernel<<<(B + 255) / 256, 256, 0, stream>>>(dep_label, B, poff, cursors, bucket);
  pack_m_kernel<<<NUM_DEP * 32, 32, 0, stream>>>(D, M16);
  wmma_tile_kernel<<<maxTiles, 32, 0, stream>>>(input_label, out_label, noise,
                                                U, V, bucket, tileDep, M16, perb);
  reduce_kernel<<<1, 256, 0, stream>>>(perb, B, (float*)d_out);
}
